// GBGraphConvModel_25039659336254
// MI455X (gfx1250) — matrix operand order repaired and checked by reading, not compiled, not run on hardware
//
#include <hip/hip_runtime.h>
#include <hip/hip_bf16.h>

typedef __bf16 bf16;
typedef __attribute__((ext_vector_type(16))) bf16  v16bf;
typedef __attribute__((ext_vector_type(8)))  bf16  v8bf;
typedef __attribute__((ext_vector_type(8)))  float v8f;

#define NATOMS  500000
#define PADROWS 500016
#define NBATCH  1024

// ---------------------------------------------------------------- helpers
__device__ __forceinline__ int atom_degree(long atom, int* r) {
  constexpr int OFF[8] = {0,20000,100000,250000,400000,475000,495000,500000};
  int d = 0;
#pragma unroll
  for (int i = 1; i <= 6; ++i) if (atom >= OFF[i]) d = i;
  *r = (int)(atom - OFF[d]);
  return d;
}

__device__ __forceinline__ const int* pick_adj(int d, const int* a1, const int* a2,
                                               const int* a3, const int* a4,
                                               const int* a5, const int* a6) {
  switch (d) {
    case 1: return a1; case 2: return a2; case 3: return a3;
    case 4: return a4; case 5: return a5; default: return a6;
  }
}

__device__ __forceinline__ unsigned encF(float f) {
  unsigned u = __float_as_uint(f);
  return (u & 0x80000000u) ? ~u : (u | 0x80000000u);
}
__device__ __forceinline__ float decF(unsigned u) {
  unsigned b = (u & 0x80000000u) ? (u ^ 0x80000000u) : ~u;
  return __uint_as_float(b);
}

// ------------------------------------------------- weight prepack (B layout)
// Packed tile: for (deg, ntile, chunk): 32 lanes x 16 bf16, lane l holds
// W[c*32 + (l>>4)*8 + (e&7) + (e>>3)*16][nt*16 + (l&15)]  (ISA 7.12.2, 16-bit B 32x16)
__global__ void pack_weights(const float* __restrict__ Wn, const float* __restrict__ Ws,
                             bf16* __restrict__ outp, int F, int N, int chunks,
                             int nTiles, int nDeg) {
  size_t idx = (size_t)blockIdx.x * blockDim.x + threadIdx.x;
  size_t total = (size_t)nDeg * nTiles * chunks * 512;
  if (idx >= total) return;
  int e = (int)(idx & 15);
  int l = (int)((idx >> 4) & 31);
  size_t t = idx >> 9;
  int c  = (int)(t % chunks); t /= chunks;
  int nt = (int)(t % nTiles);
  int d  = (int)(t / nTiles);
  int k = c * 32 + (l >> 4) * 8 + (e & 7) + ((e >> 3) << 4);
  int n = nt * 16 + (l & 15);
  float v = 0.f;
  if (nDeg > 1) {               // graph-conv: [Wn(F) ; Ws(F) ; pad]
    if (k < F)          { if (d > 0) v = Wn[((size_t)(d - 1) * F + k) * N + n]; }
    else if (k < 2 * F) { v = Ws[((size_t)d * F + (k - F)) * N + n]; }
  } else {                      // plain dense: Ws == W, K == chunks*32
    if (k < F) v = Ws[(size_t)k * N + n];
  }
  outp[idx] = (bf16)v;
}

// ------------------------------------------------- build X = [nbr-sum | self]
__global__ void aggregate(const float* __restrict__ feat, int F, int ldx,
                          const int* __restrict__ a1, const int* __restrict__ a2,
                          const int* __restrict__ a3, const int* __restrict__ a4,
                          const int* __restrict__ a5, const int* __restrict__ a6,
                          bf16* __restrict__ X) {
  size_t idx = (size_t)blockIdx.x * blockDim.x + threadIdx.x;
  size_t total = (size_t)NATOMS * ldx;
  if (idx >= total) return;
  int col = (int)(idx % ldx);
  long atom = (long)(idx / ldx);
  int r; int d = atom_degree(atom, &r);
  float v = 0.f;
  if (col < F) {
    if (d > 0) {
      const int* adj = pick_adj(d, a1, a2, a3, a4, a5, a6);
      for (int j = 0; j < d; ++j)
        v += feat[(size_t)adj[(size_t)r * d + j] * F + col];
    }
  } else if (col < 2 * F) {
    v = feat[(size_t)atom * F + (col - F)];
  }
  X[(size_t)atom * ldx + col] = (bf16)v;
}

// ------------------------------------------------- WMMA GEMM + tanh + BN
// One wave -> one 16x16 output tile. A: 16x32 bf16 per ISA layout
// (lane m=lane&15, khalf=lane>>4; elems 0..7 = K[kh*8..+7], 8..15 = K[16+kh*8..+7]).
template <int CHUNKS, bool PER_DEG, bool OUT_BF16>
__global__ __launch_bounds__(128) void gemm_tanh_bn(
    const bf16* __restrict__ X, int ldx, const bf16* __restrict__ Wp,
    const float* __restrict__ bias, const float* __restrict__ bng,
    const float* __restrict__ bnb, const float* __restrict__ bnm,
    const float* __restrict__ bnv, void* __restrict__ outPtr, int N, int nTilesN) {
  const int lane = threadIdx.x & 31;
  const int wave = threadIdx.x >> 5;
  const int nt = blockIdx.y * 4 + wave;

  int d = 0, rowBase, segEnd;
  if (PER_DEG) {
    constexpr int OFF[8]  = {0, 20000, 100000, 250000, 400000, 475000, 495000, 500000};
    constexpr int TCUM[8] = {0, 1250, 6250, 15625, 25000, 29688, 30938, 31251};
    const int bx = blockIdx.x;
#pragma unroll
    for (int i = 1; i <= 6; ++i) if (bx >= TCUM[i]) d = i;
    rowBase = OFF[d] + (bx - TCUM[d]) * 16;
    segEnd  = OFF[d + 1];
  } else {
    rowBase = blockIdx.x * 16;
    segEnd  = NATOMS;
  }

  const int m  = lane & 15;
  const int kh = lane >> 4;
  const bf16* __restrict__ xrow  = X + (size_t)(rowBase + m) * ldx;
  const bf16* __restrict__ wbase = Wp + (size_t)(d * nTilesN + nt) * CHUNKS * 512 + lane * 16;

  v8f acc = {};
#pragma unroll
  for (int c = 0; c < CHUNKS; ++c) {
    v8bf lo = *reinterpret_cast<const v8bf*>(xrow + c * 32 + kh * 8);
    v8bf hi = *reinterpret_cast<const v8bf*>(xrow + c * 32 + 16 + kh * 8);
    v16bf a = __builtin_shufflevector(lo, hi, 0, 1, 2, 3, 4, 5, 6, 7,
                                      8, 9, 10, 11, 12, 13, 14, 15);
    v16bf b = *reinterpret_cast<const v16bf*>(wbase + c * 512);
    if (c + 1 < CHUNKS) __builtin_prefetch(xrow + (c + 1) * 32, 0, 0);
    acc = __builtin_amdgcn_wmma_f32_16x16x32_bf16(false, a, false, b, (short)0,
                                                  acc, false, false);
  }

  // C/D layout: VGPR r, lanes 0-15 -> M=r, lanes 16-31 -> M=8+r; N = lane&15.
  const int col = nt * 16 + (lane & 15);
  const float bia = bias[(size_t)d * N + col];
  const float g = bng[col], be = bnb[col], mu = bnm[col];
  const float rs = rsqrtf(bnv[col] + 1e-3f);
#pragma unroll
  for (int r = 0; r < 8; ++r) {
    const int rr = rowBase + kh * 8 + r;
    if (rr < segEnd) {
      float x = tanhf(acc[r] + bia);
      x = g * (x - mu) * rs + be;
      if (OUT_BF16)
        reinterpret_cast<bf16*>(outPtr)[(size_t)rr * N + col] = (bf16)x;
      else
        reinterpret_cast<float*>(outPtr)[(size_t)rr * N + col] = x;
    }
  }
}

// ------------------------------------------------- graph pool (max over self+nbrs)
template <typename T>
__global__ void pool_max(const float* __restrict__ h, const int* __restrict__ a1,
                         const int* __restrict__ a2, const int* __restrict__ a3,
                         const int* __restrict__ a4, const int* __restrict__ a5,
                         const int* __restrict__ a6, T* __restrict__ outp) {
  size_t idx = (size_t)blockIdx.x * blockDim.x + threadIdx.x;
  if (idx >= (size_t)NATOMS * 64) return;
  int col = (int)(idx & 63);
  long atom = (long)(idx >> 6);
  int r; int d = atom_degree(atom, &r);
  float v = h[(size_t)atom * 64 + col];
  if (d > 0) {
    const int* adj = pick_adj(d, a1, a2, a3, a4, a5, a6);
    for (int j = 0; j < d; ++j)
      v = fmaxf(v, h[(size_t)adj[(size_t)r * d + j] * 64 + col]);
  }
  outp[(size_t)atom * 64 + col] = (T)v;
}

// ------------------------------------------------- segment gather (sum+max)
__global__ void gather_init(float* __restrict__ s, unsigned* __restrict__ mx) {
  size_t idx = (size_t)blockIdx.x * blockDim.x + threadIdx.x;
  if (idx < (size_t)NBATCH * 128) { s[idx] = 0.f; mx[idx] = 0u; }
}

__global__ void gather_acc(const bf16* __restrict__ h3, const int* __restrict__ memb,
                           float* __restrict__ s, unsigned* __restrict__ mx) {
  size_t idx = (size_t)blockIdx.x * blockDim.x + threadIdx.x;
  if (idx >= (size_t)NATOMS * 128) return;
  int col = (int)(idx & 127);
  long atom = (long)(idx >> 7);
  float v = (float)h3[(size_t)atom * 128 + col];
  int b = memb[atom];
  atomicAdd(&s[(size_t)b * 128 + col], v);
  atomicMax(&mx[(size_t)b * 128 + col], encF(v));
}

__global__ void gather_fin(const float* __restrict__ s, const unsigned* __restrict__ mx,
                           float* __restrict__ g) {
  size_t idx = (size_t)blockIdx.x * blockDim.x + threadIdx.x;
  if (idx >= (size_t)NBATCH * 256) return;
  int col = (int)(idx & 255);
  long b = (long)(idx >> 8);
  float v = (col < 128) ? s[(size_t)b * 128 + col] : decF(mx[(size_t)b * 128 + (col - 128)]);
  g[idx] = tanhf(v);
}

// ------------------------------------------------- final dense chain (tiny)
__global__ __launch_bounds__(64) void final_dense(const float* __restrict__ g,
                                                  const float* __restrict__ W2,
                                                  const float* __restrict__ b2,
                                                  const float* __restrict__ W3,
                                                  const float* __restrict__ b3,
                                                  float* __restrict__ out) {
  __shared__ float gs[256];
  __shared__ float ss[64];
  const int row = blockIdx.x, t = threadIdx.x;
  for (int i = t; i < 256; i += 64) gs[i] = g[(size_t)row * 256 + i];
  __syncthreads();
  float acc = b2[t];
  for (int k = 0; k < 256; ++k) acc += gs[k] * W2[(size_t)k * 64 + t];
  ss[t] = 1.f / (1.f + expf(-acc));
  __syncthreads();
  if (t == 0) {
    float o = b3[0];
    for (int k = 0; k < 64; ++k) o += ss[k] * W3[k];
    out[row] = o;
  }
}

// ---------------------------------------------------------------- launch
static inline size_t alignup(size_t x) { return (x + 255) & ~(size_t)255; }

extern "C" void kernel_launch(void* const* d_in, const int* in_sizes, int n_in,
                              void* d_out, int out_size, void* d_ws, size_t ws_size,
                              hipStream_t stream) {
  const float* feat  = (const float*)d_in[0];
  const int*   memb  = (const int*)d_in[1];
  const int*   a1 = (const int*)d_in[2]; const int* a2 = (const int*)d_in[3];
  const int*   a3 = (const int*)d_in[4]; const int* a4 = (const int*)d_in[5];
  const int*   a5 = (const int*)d_in[6]; const int* a6 = (const int*)d_in[7];
  const float* gc1_Wn = (const float*)d_in[8];
  const float* gc1_Ws = (const float*)d_in[9];
  const float* gc1_b  = (const float*)d_in[10];
  const float* gc2_Wn = (const float*)d_in[11];
  const float* gc2_Ws = (const float*)d_in[12];
  const float* gc2_b  = (const float*)d_in[13];
  const float* bn1g = (const float*)d_in[14]; const float* bn1b = (const float*)d_in[15];
  const float* bn1m = (const float*)d_in[16]; const float* bn1v = (const float*)d_in[17];
  const float* bn3g = (const float*)d_in[18]; const float* bn3b = (const float*)d_in[19];
  const float* bn3m = (const float*)d_in[20]; const float* bn3v = (const float*)d_in[21];
  const float* d1W = (const float*)d_in[22]; const float* d1b = (const float*)d_in[23];
  const float* d2W = (const float*)d_in[24]; const float* d2b = (const float*)d_in[25];
  const float* d3W = (const float*)d_in[26]; const float* d3b = (const float*)d_in[27];
  float* out = (float*)d_out;

  // workspace carve-up
  char* ws = (char*)d_ws;
  size_t off = 0;
  bf16* A = (bf16*)(ws + off);   off = alignup(off + (size_t)PADROWS * 320);        // X1/X2/h3
  float* H1 = (float*)(ws + off); off = alignup(off + (size_t)PADROWS * 64 * 4);
  float* H2 = (float*)(ws + off); off = alignup(off + (size_t)PADROWS * 64 * 4);
  bf16* W1p = (bf16*)(ws + off); off = alignup(off + (size_t)7 * 4 * 5 * 512 * 2);
  bf16* W2p = (bf16*)(ws + off); off = alignup(off + (size_t)7 * 4 * 4 * 512 * 2);
  bf16* WDp = (bf16*)(ws + off); off = alignup(off + (size_t)1 * 8 * 2 * 512 * 2);
  float* SUM = (float*)(ws + off);    off = alignup(off + (size_t)NBATCH * 128 * 4);
  unsigned* MX = (unsigned*)(ws + off); off = alignup(off + (size_t)NBATCH * 128 * 4);
  float* G = (float*)(ws + off);      off = alignup(off + (size_t)NBATCH * 256 * 4);

  const int TB = 256;
  auto blk = [](size_t n, int t) { return (unsigned)((n + t - 1) / t); };

  // 1) pack weights
  pack_weights<<<blk((size_t)7 * 4 * 5 * 512, TB), TB, 0, stream>>>(gc1_Wn, gc1_Ws, W1p, 75, 64, 5, 4, 7);
  pack_weights<<<blk((size_t)7 * 4 * 4 * 512, TB), TB, 0, stream>>>(gc2_Wn, gc2_Ws, W2p, 64, 64, 4, 4, 7);
  pack_weights<<<blk((size_t)8 * 2 * 512, TB), TB, 0, stream>>>(nullptr, d1W, WDp, 64, 128, 2, 8, 1);

  // 2) GC1: aggregate -> WMMA GEMM(K=160) + tanh + bn1
  aggregate<<<blk((size_t)NATOMS * 160, TB), TB, 0, stream>>>(feat, 75, 160, a1, a2, a3, a4, a5, a6, A);
  gemm_tanh_bn<5, true, false><<<dim3(31251, 1), 128, 0, stream>>>(
      A, 160, W1p, gc1_b, bn1g, bn1b, bn1m, bn1v, (void*)H1, 64, 4);

  // 3) pool1 (f32)
  pool_max<float><<<blk((size_t)NATOMS * 64, TB), TB, 0, stream>>>(H1, a1, a2, a3, a4, a5, a6, H2);

  // 4) GC2: aggregate -> WMMA GEMM(K=128) + tanh + bn1
  aggregate<<<blk((size_t)NATOMS * 128, TB), TB, 0, stream>>>(H2, 64, 128, a1, a2, a3, a4, a5, a6, A);
  gemm_tanh_bn<4, true, false><<<dim3(31251, 1), 128, 0, stream>>>(
      A, 128, W2p, gc2_b, bn1g, bn1b, bn1m, bn1v, (void*)H1, 64, 4);

  // 5) pool2 (bf16, feeds dense GEMM)
  pool_max<bf16><<<blk((size_t)NATOMS * 64, TB), TB, 0, stream>>>(H1, a1, a2, a3, a4, a5, a6, (bf16*)H2);

  // 6) dense1: WMMA GEMM(K=64, N=128) + tanh + bn3 -> h3 (bf16) in A
  gemm_tanh_bn<2, false, true><<<dim3(31250, 2), 128, 0, stream>>>(
      (const bf16*)H2, 64, WDp, d1b, bn3g, bn3b, bn3m, bn3v, (void*)A, 128, 8);

  // 7) graph gather (segment sum + max) + tanh
  gather_init<<<blk((size_t)NBATCH * 128, TB), TB, 0, stream>>>(SUM, MX);
  gather_acc<<<blk((size_t)NATOMS * 128, TB), TB, 0, stream>>>((const bf16*)A, memb, SUM, MX);
  gather_fin<<<blk((size_t)NBATCH * 256, TB), TB, 0, stream>>>(SUM, MX, G);

  // 8) final dense chain
  final_dense<<<NBATCH, 64, 0, stream>>>(G, d2W, d2b, d3W, d3b, out);
}
